// MaskYolo_43155831390679
// MI455X (gfx1250) — matrix operand order repaired and checked by reading, not compiled, not run on hardware
//
#include <hip/hip_runtime.h>
#include <hip/hip_bf16.h>

typedef __attribute__((ext_vector_type(16))) __bf16 v16bf;
typedef __attribute__((ext_vector_type(8)))  float  v8f;

union BfTile {
    v16bf v;
    uint4 q[2];
    unsigned short s[16];
};

__device__ __forceinline__ unsigned short f32_to_bf16(float f) {
    union { float f; unsigned u; } v; v.f = f;
    unsigned r = v.u + 0x7FFFu + ((v.u >> 16) & 1u);   // round-to-nearest-even
    return (unsigned short)(r >> 16);
}
__device__ __forceinline__ float bf16_to_f32(unsigned short s) {
    union { unsigned u; float f; } v; v.u = ((unsigned)s) << 16;
    return v.f;
}

// ---------------------------------------------------------------------------
// Weight repack: w[O][I][KS][KS] fp32 (OIHW)  ->  wt[O][tap*I + ic] bf16
// k-order = (ky,kx,ic) so activation NHWC im2col chunks are contiguous.
// ---------------------------------------------------------------------------
__global__ __launch_bounds__(256)
void convert_w(const float* __restrict__ w, unsigned short* __restrict__ wt,
               int O, int I, int KS) {
    int total = O * I * KS * KS;
    int t = blockIdx.x * blockDim.x + threadIdx.x;
    if (t >= total) return;
    int K  = KS * KS * I;
    int o  = t / K;
    int k  = t - o * K;
    int tap = k / I;
    int ic  = k - tap * I;
    int ky  = tap / KS;
    int kx  = tap - ky * KS;
    wt[t] = f32_to_bf16(w[(((size_t)o * I + ic) * KS + ky) * KS + kx]);
}

// NCHW fp32 -> NHWC bf16
__global__ __launch_bounds__(256)
void pack_nhwc(const float* __restrict__ in, unsigned short* __restrict__ out,
               int B, int C, int H, int W) {
    int total = B * C * H * W;
    int t = blockIdx.x * blockDim.x + threadIdx.x;
    if (t >= total) return;
    int c = t % C;
    int x = (t / C) % W;
    int y = (t / (C * W)) % H;
    int b = t / (C * W * H);
    out[t] = f32_to_bf16(in[(((size_t)b * C + c) * H + y) * W + x]);
}

// ---------------------------------------------------------------------------
// ROI-align over the virtual 960-ch merged pyramid (nearest-resize folded in).
// Writes NHWC bf16 [12544][960].
// ---------------------------------------------------------------------------
__global__ __launch_bounds__(256)
void roi_align_pack(const float* __restrict__ f0, const float* __restrict__ f1,
                    const float* __restrict__ f2, const float* __restrict__ f3,
                    const float* __restrict__ boxes, unsigned short* __restrict__ outA) {
    int samp = blockIdx.x;                 // 0..12543
    int r  = samp / 196;
    int p  = samp - r * 196;
    int sy = p / 14, sx = p - sy * 14;
    int b  = r >> 4;

    float x1 = boxes[r * 4 + 0], y1 = boxes[r * 4 + 1];
    float x2 = boxes[r * 4 + 2], y2 = boxes[r * 4 + 3];
    float xg = x1 + ((float)sx + 0.5f) * (x2 - x1) * (1.0f / 14.0f);
    float yg = y1 + ((float)sy + 0.5f) * (y2 - y1) * (1.0f / 14.0f);
    bool valid = (xg >= -1.0f) && (xg <= 128.0f) && (yg >= -1.0f) && (yg <= 128.0f);
    xg = fminf(fmaxf(xg, 0.0f), 127.0f);
    yg = fminf(fmaxf(yg, 0.0f), 127.0f);
    int x0 = (int)floorf(xg), y0 = (int)floorf(yg);
    int x1i = min(x0 + 1, 127), y1i = min(y0 + 1, 127);
    float lx = xg - (float)x0, ly = yg - (float)y0;
    float w00 = (1.f - ly) * (1.f - lx), w01 = (1.f - ly) * lx;
    float w10 = ly * (1.f - lx),          w11 = ly * lx;

    unsigned short* orow = outA + (size_t)samp * 960;
    for (int c = threadIdx.x; c < 960; c += blockDim.x) {
        const float* src; int cc, Hk, Ck, sh;
        if (c < 64)       { src = f0; cc = c;       Hk = 128; Ck = 64;  sh = 0; }
        else if (c < 192) { src = f1; cc = c - 64;  Hk = 64;  Ck = 128; sh = 1; }
        else if (c < 448) { src = f2; cc = c - 192; Hk = 32;  Ck = 256; sh = 2; }
        else              { src = f3; cc = c - 448; Hk = 16;  Ck = 512; sh = 3; }
        int Y0 = y0 >> sh, Y1 = y1i >> sh, X0 = x0 >> sh, X1 = x1i >> sh;
        const float* base = src + ((size_t)(b * Ck + cc) * Hk) * Hk;
        float v = w00 * base[Y0 * Hk + X0] + w01 * base[Y0 * Hk + X1] +
                  w10 * base[Y1 * Hk + X0] + w11 * base[Y1 * Hk + X1];
        orow[c] = f32_to_bf16(valid ? v : 0.0f);
    }
}

// ---------------------------------------------------------------------------
// Implicit-GEMM conv, fully specialized: fused bias + SiLU, bf16 in/out.
// Per wave: 32M x 64N tile; K-loop step 32 with v_wmma_f32_16x16x32_bf16,
// manual 2-stage software pipeline (all KTOT/32 counts are even).
// Out-of-bounds 3x3 taps read a zeroed row (zrow) -> no masking, no branches.
// A: act NHWC [pixel][CIN]; B: wt [N][KS*KS*CIN]. CIN % 32 == 0,
// grid = (N/64, M/128), M % 128 == 0.
// ---------------------------------------------------------------------------
template<int KS, int PAD, int CIN, int N, int H, int W>
__global__ __launch_bounds__(128)
void wmma_conv_gemm(const unsigned short* __restrict__ A,
                    const unsigned short* __restrict__ Wt,
                    const float* __restrict__ bias,
                    unsigned short* __restrict__ Out,
                    const unsigned short* __restrict__ zrow) {
    constexpr int KTOT = KS * KS * CIN;
    constexpr int NKB  = KTOT / 32;        // even for all instantiations
    const int lane = threadIdx.x & 31;
    const int wave = threadIdx.x >> 5;
    const int l15  = lane & 15;
    const int hi   = lane >> 4;            // 0: lanes 0-15, 1: lanes 16-31
    const int n0   = blockIdx.x * 64;
    const int mb   = blockIdx.y * 128 + wave * 32;
    const int c0   = hi * 8;               // A K-chunk start within 32-block

    int mrow[2], ry[2], rx[2];
#pragma unroll
    for (int ms = 0; ms < 2; ++ms) {
        mrow[ms] = mb + ms * 16 + l15;
        rx[ms] = mrow[ms] % W;
        ry[ms] = (mrow[ms] / W) % H;
    }

    const unsigned short* bp[4];
#pragma unroll
    for (int j = 0; j < 4; ++j)
        bp[j] = Wt + (size_t)(n0 + j * 16 + l15) * KTOT + hi * 8;

    v8f acc[2][4];
#pragma unroll
    for (int i = 0; i < 2; ++i)
#pragma unroll
        for (int j = 0; j < 4; ++j)
#pragma unroll
            for (int k = 0; k < 8; ++k) acc[i][j][k] = 0.0f;

    // per-tap A row pointers (OOB taps -> zeroed row)
    const unsigned short* ar[2];
    auto setTap = [&](int tap) {
#pragma unroll
        for (int ms = 0; ms < 2; ++ms) {
            if (KS == 1) {
                ar[ms] = A + (size_t)mrow[ms] * CIN + c0;
            } else {
                int dy = tap / KS - PAD;
                int dx = tap - (tap / KS) * KS - PAD;
                int ys = ry[ms] + dy, xs = rx[ms] + dx;
                bool ok = ((unsigned)ys < (unsigned)H) && ((unsigned)xs < (unsigned)W);
                ar[ms] = ok ? (A + (size_t)(mrow[ms] + dy * W + dx) * CIN + c0)
                            : (zrow + c0);
            }
        }
    };

    int tap = 0, icb = 0;
    auto advance = [&]() {
        icb += 32;
        if (icb == CIN) { icb = 0; ++tap; setTap(tap); }
    };

    BfTile a0[2], b0[4], a1[2], b1[4];
    auto loadTiles = [&](BfTile (&a)[2], BfTile (&b)[4], int icb_, int kb_) {
#pragma unroll
        for (int ms = 0; ms < 2; ++ms) {
            const unsigned short* ap = ar[ms] + icb_;
            a[ms].q[0] = *reinterpret_cast<const uint4*>(ap);       // K c0..c0+7
            a[ms].q[1] = *reinterpret_cast<const uint4*>(ap + 16);  // K c0+16..23
        }
#pragma unroll
        for (int j = 0; j < 4; ++j) {
            const unsigned short* p = bp[j] + kb_;
            b[j].q[0] = *reinterpret_cast<const uint4*>(p);         // K hi*8+0..7
            b[j].q[1] = *reinterpret_cast<const uint4*>(p + 16);    // K hi*8+16..23
        }
    };
    auto compute = [&](BfTile (&a)[2], BfTile (&b)[4]) {
#pragma unroll
        for (int j = 0; j < 4; ++j) {
            acc[0][j] = __builtin_amdgcn_wmma_f32_16x16x32_bf16(
                false, a[0].v, false, b[j].v, (short)0, acc[0][j], false, false);
            acc[1][j] = __builtin_amdgcn_wmma_f32_16x16x32_bf16(
                false, a[1].v, false, b[j].v, (short)0, acc[1][j], false, false);
        }
    };

    setTap(0);
    loadTiles(a0, b0, 0, 0);
    for (int it = 0; it < NKB / 2; ++it) {
        int base = it * 64;
        advance();                               // -> base+32 (always < KTOT)
        loadTiles(a1, b1, icb, base + 32);
        compute(a0, b0);
        if (it + 1 < NKB / 2) {
            advance();                           // -> base+64
            loadTiles(a0, b0, icb, base + 64);
        }
        compute(a1, b1);
    }

    // epilogue: bias + SiLU, bf16 store. C layout: VGPR v -> row v (+8 for hi lanes)
#pragma unroll
    for (int ms = 0; ms < 2; ++ms) {
#pragma unroll
        for (int j = 0; j < 4; ++j) {
            int n = n0 + j * 16 + l15;
            float bv = bias[n];
#pragma unroll
            for (int v = 0; v < 8; ++v) {
                int m = mb + ms * 16 + v + hi * 8;
                float val = acc[ms][j][v] + bv;
                val = val / (1.0f + __expf(-val));   // SiLU
                Out[(size_t)m * N + n] = f32_to_bf16(val);
            }
        }
    }
}

// ---------------------------------------------------------------------------
// Small head outputs (4+1+80 channels) fused with YOLO decode.
// ---------------------------------------------------------------------------
__global__ __launch_bounds__(256)
void head_out_decode(const unsigned short* __restrict__ h1,
                     const unsigned short* __restrict__ h2,
                     const float* __restrict__ reg_w, const float* __restrict__ reg_b,
                     const float* __restrict__ obj_w, const float* __restrict__ obj_b,
                     const float* __restrict__ cls_w, const float* __restrict__ cls_b,
                     float* __restrict__ out,
                     int Mtot, int HW, int Wd, int lvl_off, float stride_) {
    int t = blockIdx.x * blockDim.x + threadIdx.x;
    if (t >= Mtot * 85) return;
    int m = t / 85, c = t - m * 85;
    int b = m / HW, hw = m - b * HW;
    const unsigned short* hrow;
    const float* wrow;
    float s;
    if (c < 4)       { hrow = h1 + (size_t)m * 256; wrow = reg_w + c * 256;       s = reg_b[c]; }
    else if (c == 4) { hrow = h1 + (size_t)m * 256; wrow = obj_w;                 s = obj_b[0]; }
    else             { hrow = h2 + (size_t)m * 256; wrow = cls_w + (c - 5) * 256; s = cls_b[c - 5]; }
    for (int k = 0; k < 256; ++k) s += bf16_to_f32(hrow[k]) * wrow[k];
    if (c == 0)                s = (s + (float)(hw % Wd)) * stride_;
    else if (c == 1)           s = (s + (float)(hw / Wd)) * stride_;
    else if (c == 2 || c == 3) s = __expf(s) * stride_;
    out[((size_t)b * 320 + lvl_off + hw) * 85 + c] = s;
}

// 1x1 conv 128 -> 1 for the mask output, fp32 out.
__global__ __launch_bounds__(256)
void mask_out(const unsigned short* __restrict__ h, const float* __restrict__ w,
              const float* __restrict__ bias, float* __restrict__ out, int M) {
    int t = blockIdx.x * blockDim.x + threadIdx.x;
    if (t >= M) return;
    float s = bias[0];
    const unsigned short* hrow = h + (size_t)t * 128;
    for (int k = 0; k < 128; ++k) s += bf16_to_f32(hrow[k]) * w[k];
    out[108800 + t] = s;   // after yolo block (4*320*85)
}

// ---------------------------------------------------------------------------
extern "C" void kernel_launch(void* const* d_in, const int* in_sizes, int n_in,
                              void* d_out, int out_size, void* d_ws, size_t ws_size,
                              hipStream_t stream) {
    if (n_in < 44) return;  // expect 5 feats + boxes + 38 param leaves

    const float* feat0 = (const float*)d_in[0];
    const float* feat1 = (const float*)d_in[1];
    const float* feat2 = (const float*)d_in[2];
    const float* feat3 = (const float*)d_in[3];
    const float* feat4 = (const float*)d_in[4];
    const float* boxes = (const float*)d_in[5];

    // params pytree order (alphabetical dict keys): head -> [unit0, unit1], mask.
    // per-unit leaf order: cls1_b, cls1_w, cls2_b, cls2_w, cls_out_b, cls_out_w,
    //   obj_out_b, obj_out_w, reg1_b, reg1_w, reg2_b, reg2_w, reg_out_b,
    //   reg_out_w, stem_b, stem_w
    const int U0 = 6, U1 = 22, MK = 38;
    auto P = [&](int i) { return (const float*)d_in[i]; };

    float* out = (float*)d_out;

    // -------- workspace layout (256B aligned, bf16 elements) --------
    size_t off = 0;
    auto take = [&](size_t elems) -> unsigned short* {
        unsigned short* p = (unsigned short*)((char*)d_ws + off);
        off += ((elems * 2 + 255) / 256) * 256;
        return p;
    };
    unsigned short* zrow     = take(1024);        // shared zero row for OOB taps
    // weights
    unsigned short* wt_stem0 = take(256 * 512);
    unsigned short* wt_r1_0  = take(256 * 2304);
    unsigned short* wt_r2_0  = take(256 * 2304);
    unsigned short* wt_c1_0  = take(256 * 2304);
    unsigned short* wt_c2_0  = take(256 * 2304);
    unsigned short* wt_stem1 = take(256 * 1024);
    unsigned short* wt_r1_1  = take(256 * 2304);
    unsigned short* wt_r2_1  = take(256 * 2304);
    unsigned short* wt_c1_1  = take(256 * 2304);
    unsigned short* wt_c2_1  = take(256 * 2304);
    unsigned short* wt_m1    = take(128 * 8640);
    unsigned short* wt_m2    = take(128 * 1152);
    // activations
    unsigned short* a3    = take(1024 * 512);   // feat3 NHWC bf16
    unsigned short* a4    = take(256 * 1024);   // feat4 NHWC bf16
    unsigned short* hs0   = take(1024 * 256);
    unsigned short* hr0   = take(1024 * 256);
    unsigned short* h1_0  = take(1024 * 256);
    unsigned short* hc0   = take(1024 * 256);
    unsigned short* h2_0  = take(1024 * 256);
    unsigned short* hs1   = take(256 * 256);
    unsigned short* hr1   = take(256 * 256);
    unsigned short* h1_1  = take(256 * 256);
    unsigned short* hc1   = take(256 * 256);
    unsigned short* h2_1  = take(256 * 256);
    unsigned short* roiA  = take((size_t)12544 * 960);
    unsigned short* hm1   = take((size_t)12544 * 128);
    unsigned short* hm2   = take((size_t)12544 * 128);
    if (off > ws_size) return;

    hipMemsetAsync((void*)zrow, 0, 1024 * sizeof(unsigned short), stream);

    auto convW = [&](const float* w, unsigned short* wt, int O, int I, int KS) {
        int total = O * I * KS * KS;
        convert_w<<<(total + 255) / 256, 256, 0, stream>>>(w, wt, O, I, KS);
    };
    // ---- weight conversions ----
    convW(P(U0 + 15), wt_stem0, 256, 512, 1);
    convW(P(U0 + 9),  wt_r1_0,  256, 256, 3);
    convW(P(U0 + 11), wt_r2_0,  256, 256, 3);
    convW(P(U0 + 1),  wt_c1_0,  256, 256, 3);
    convW(P(U0 + 3),  wt_c2_0,  256, 256, 3);
    convW(P(U1 + 15), wt_stem1, 256, 1024, 1);
    convW(P(U1 + 9),  wt_r1_1,  256, 256, 3);
    convW(P(U1 + 11), wt_r2_1,  256, 256, 3);
    convW(P(U1 + 1),  wt_c1_1,  256, 256, 3);
    convW(P(U1 + 3),  wt_c2_1,  256, 256, 3);
    convW(P(MK + 1),  wt_m1,    128, 960, 3);
    convW(P(MK + 3),  wt_m2,    128, 128, 3);

    // ---- activation packing ----
    pack_nhwc<<<(4 * 512 * 256 + 255) / 256, 256, 0, stream>>>(feat3, a3, 4, 512, 16, 16);
    pack_nhwc<<<(4 * 1024 * 64 + 255) / 256, 256, 0, stream>>>(feat4, a4, 4, 1024, 8, 8);

    // ---- ROI-align (merged pyramid, nearest-resize folded into gather) ----
    roi_align_pack<<<12544, 256, 0, stream>>>(feat0, feat1, feat2, feat3, boxes, roiA);

    // ---- head level 0 (feat3, stride 32): M = 4*16*16 = 1024 ----
    {
        dim3 g(4, 8);
        wmma_conv_gemm<1, 0, 512, 256, 16, 16><<<g, 128, 0, stream>>>(a3,  wt_stem0, P(U0 + 14), hs0,  zrow);
        wmma_conv_gemm<3, 1, 256, 256, 16, 16><<<g, 128, 0, stream>>>(hs0, wt_r1_0,  P(U0 + 8),  hr0,  zrow);
        wmma_conv_gemm<3, 1, 256, 256, 16, 16><<<g, 128, 0, stream>>>(hr0, wt_r2_0,  P(U0 + 10), h1_0, zrow);
        wmma_conv_gemm<3, 1, 256, 256, 16, 16><<<g, 128, 0, stream>>>(hs0, wt_c1_0,  P(U0 + 0),  hc0,  zrow);
        wmma_conv_gemm<3, 1, 256, 256, 16, 16><<<g, 128, 0, stream>>>(hc0, wt_c2_0,  P(U0 + 2),  h2_0, zrow);
    }
    // ---- head level 1 (feat4, stride 64): M = 4*8*8 = 256 ----
    {
        dim3 g(4, 2);
        wmma_conv_gemm<1, 0, 1024, 256, 8, 8><<<g, 128, 0, stream>>>(a4,  wt_stem1, P(U1 + 14), hs1,  zrow);
        wmma_conv_gemm<3, 1, 256,  256, 8, 8><<<g, 128, 0, stream>>>(hs1, wt_r1_1,  P(U1 + 8),  hr1,  zrow);
        wmma_conv_gemm<3, 1, 256,  256, 8, 8><<<g, 128, 0, stream>>>(hr1, wt_r2_1,  P(U1 + 10), h1_1, zrow);
        wmma_conv_gemm<3, 1, 256,  256, 8, 8><<<g, 128, 0, stream>>>(hs1, wt_c1_1,  P(U1 + 0),  hc1,  zrow);
        wmma_conv_gemm<3, 1, 256,  256, 8, 8><<<g, 128, 0, stream>>>(hc1, wt_c2_1,  P(U1 + 2),  h2_1, zrow);
    }

    // ---- head outputs + decode ----
    head_out_decode<<<(1024 * 85 + 255) / 256, 256, 0, stream>>>(
        h1_0, h2_0, P(U0 + 13), P(U0 + 12), P(U0 + 7), P(U0 + 6),
        P(U0 + 5), P(U0 + 4), out, 1024, 256, 16, 0, 32.0f);
    head_out_decode<<<(256 * 85 + 255) / 256, 256, 0, stream>>>(
        h1_1, h2_1, P(U1 + 13), P(U1 + 12), P(U1 + 7), P(U1 + 6),
        P(U1 + 5), P(U1 + 4), out, 256, 64, 8, 256, 64.0f);

    // ---- mask branch: M = 64*14*14 = 12544 ----
    {
        dim3 g(2, 98);
        wmma_conv_gemm<3, 1, 960, 128, 14, 14><<<g, 128, 0, stream>>>(roiA, wt_m1, P(MK + 0), hm1, zrow);
        wmma_conv_gemm<3, 1, 128, 128, 14, 14><<<g, 128, 0, stream>>>(hm1,  wt_m2, P(MK + 2), hm2, zrow);
    }
    mask_out<<<(12544 + 255) / 256, 256, 0, stream>>>(hm2, P(MK + 5), P(MK + 4), out, 12544);
    (void)in_sizes; (void)out_size;
}
